// TBGAT_29978871726249
// MI455X (gfx1250) — matrix operand order repaired and checked by reading, compiled, not run on hardware
//
#include <hip/hip_runtime.h>
#include <hip/hip_bf16.h>

#define NEG_SLOPE 0.2f

typedef __attribute__((ext_vector_type(16))) _Float16 v16h;
typedef __attribute__((ext_vector_type(8)))  _Float16 v8h;
typedef __attribute__((ext_vector_type(8)))  float    v8f;

__device__ __forceinline__ void atomic_max_f32(float* addr, float val) {
  // max over mixed-sign floats via int/uint monotonicity; mbuf pre-init to -inf
  if (val >= 0.0f) atomicMax((int*)addr, __float_as_int(val));
  else             atomicMin((unsigned int*)addr, __float_as_uint(val));
}

__global__ void k_zero_f32(float* __restrict__ p, int n) {
  int i = blockIdx.x * blockDim.x + threadIdx.x;
  if (i < n) p[i] = 0.0f;
}

// Layer-1 dense transform with input column selection: H[n,m] = sum_k x[n,sel_k]*W[k,m]
__global__ void k_dense3(const float* __restrict__ x, const float* __restrict__ W,
                         float* __restrict__ H, int N, int c0, int c1, int c2) {
  int idx = blockIdx.x * blockDim.x + threadIdx.x;
  if (idx >= N * 128) return;
  int n = idx >> 7, m = idx & 127;
  const float* xr = x + (size_t)n * 5;
  H[idx] = xr[c0] * W[m] + xr[c1] * W[128 + m] + xr[c2] * W[256 + m];
}

// Pack W[K,M] (f32, row-major) into WMMA B-fragment lane order as f16.
// Fragment (mt,kb): lane holds 16 halves; lane&15 = output col, lane>>4 selects K half.
__global__ void k_pack_w(const float* __restrict__ W, _Float16* __restrict__ Wp,
                         int K, int M) {
  int kblks = K >> 5, mtiles = M >> 4;
  int idx = blockIdx.x * blockDim.x + threadIdx.x;   // (mt*kblks + kb)*32 + lane
  if (idx >= mtiles * kblks * 32) return;
  int lane = idx & 31;
  int kb = (idx >> 5) % kblks;
  int mt = (idx >> 5) / kblks;
  int hi = lane >> 4, nc = lane & 15;
  _Float16* dst = Wp + (size_t)idx * 16;
  const float* src = W + (size_t)(kb << 5) * M + (mt << 4) + nc;
#pragma unroll
  for (int t = 0; t < 16; ++t) dst[t] = (_Float16)src[(size_t)(((t >> 3) << 4) + (hi << 3) + (t & 7)) * M];
}

// OUT[N,M] = A16[N,K] @ Wpacked  via v_wmma_f32_16x16x32_f16 (f32 accumulate).
// One wave per 16x16 output tile. rowTiles*16 == N (N=50000 = 3125*16).
__global__ void k_gemm_wmma(const _Float16* __restrict__ Ah,
                            const _Float16* __restrict__ Wp,
                            float* __restrict__ OUT,
                            int rowTiles, int K, int M) {
  int gwave = (int)((blockIdx.x * blockDim.x + threadIdx.x) >> 5);
  int lane  = threadIdx.x & 31;
  int mtiles = M >> 4, kblks = K >> 5;
  int rt = gwave / mtiles, ct = gwave - rt * mtiles;
  if (rt >= rowTiles) return;          // wave-uniform guard, EXEC stays all-ones
  int r = lane & 15, hi = lane >> 4;
  const _Float16* arow = Ah + (size_t)((rt << 4) + r) * K;
  const _Float16* bp   = Wp + ((size_t)ct * kblks * 32 + lane) * 16;
  v8f acc = {};
  for (int kb = 0; kb < kblks; ++kb) {
    // A fragment: lane(hi=0): K {kk..kk+7, kk+16..23}; lane(hi=1): {kk+8..15, kk+24..31}
    v8h a0 = *(const v8h*)(arow + (kb << 5) + (hi << 3));
    v8h a1 = *(const v8h*)(arow + (kb << 5) + 16 + (hi << 3));
    v16h a;
#pragma unroll
    for (int t = 0; t < 8; ++t) { a[t] = a0[t]; a[t + 8] = a1[t]; }
    v16h b = *(const v16h*)(bp + (size_t)kb * 32 * 16);
    acc = __builtin_amdgcn_wmma_f32_16x16x32_f16(false, a, false, b,
                                                 (short)0, acc, false, false);
  }
  // C/D layout: VGPR v -> row (rt*16 + v + 8*hi), col = ct*16 + (lane&15)
  float* op = OUT + ((size_t)(rt << 4) + (hi << 3)) * M + (ct << 4) + r;
#pragma unroll
  for (int v = 0; v < 8; ++v) op[(size_t)v * M] = acc[v];
}

// Per-(node,head): attention dot products + init segment max/sum accumulators.
__global__ void k_node_prep(const float* __restrict__ H,
                            const float* __restrict__ att_s, const float* __restrict__ att_d,
                            float* __restrict__ a_s, float* __restrict__ a_d,
                            float* __restrict__ mbuf, float* __restrict__ sbuf,
                            int N, int Hh, int C) {
  int idx = blockIdx.x * blockDim.x + threadIdx.x;   // n*Hh + h
  if (idx >= N * Hh) return;
  int h = idx % Hh;
  const float* hp = H + (size_t)idx * C;             // [n,h,c] contiguous
  float as = 0.f, ad = 0.f;
  for (int c = 0; c < C; ++c) { float v = hp[c]; as += v * att_s[h * C + c]; ad += v * att_d[h * C + c]; }
  a_s[idx] = as; a_d[idx] = ad;
  mbuf[idx] = -__builtin_inff();
  sbuf[idx] = 0.f;
}

// Pass 1: leaky-relu logits + segment max (edges e<E from arrays; e>=E self loops).
__global__ void k_edge_logits(const int* __restrict__ jidx, const int* __restrict__ iidx,
                              int E, int N, int Hh,
                              const float* __restrict__ a_s, const float* __restrict__ a_d,
                              float* __restrict__ earr, float* __restrict__ mbuf) {
  int idx = blockIdx.x * blockDim.x + threadIdx.x;
  if (idx >= (E + N) * Hh) return;
  int e = idx / Hh, h = idx - e * Hh;
  int j, i;
  if (e < E) { j = jidx[e]; i = iidx[e]; } else { j = i = e - E; }
  float v = a_s[j * Hh + h] + a_d[i * Hh + h];
  v = (v > 0.f) ? v : NEG_SLOPE * v;
  earr[idx] = v;
  atomic_max_f32(&mbuf[i * Hh + h], v);
}

// Pass 2: exp(e - max) + segment sum.
__global__ void k_edge_expsum(const int* __restrict__ iidx, int E, int N, int Hh,
                              const float* __restrict__ mbuf,
                              float* __restrict__ earr, float* __restrict__ sbuf) {
  int idx = blockIdx.x * blockDim.x + threadIdx.x;
  if (idx >= (E + N) * Hh) return;
  int e = idx / Hh, h = idx - e * Hh;
  int i = (e < E) ? iidx[e] : (e - E);
  float w = __expf(earr[idx] - mbuf[i * Hh + h]);
  earr[idx] = w;
  atomicAdd(&sbuf[i * Hh + h], w);
}

// Pass 3: out[i] += alpha * h[j]. blockDim.x = Hh*C (coalesced message row), blockDim.y edges/block.
__global__ void k_edge_aggregate(const int* __restrict__ jidx, const int* __restrict__ iidx,
                                 int E, int N, int Hh, int C,
                                 const float* __restrict__ Hf, const float* __restrict__ earr,
                                 const float* __restrict__ sbuf, float* __restrict__ outacc) {
  int e = blockIdx.x * blockDim.y + threadIdx.y;
  if (e >= E + N) return;
  int t = threadIdx.x;
  int h = t / C;
  int j, i;
  if (e < E) { j = jidx[e]; i = iidx[e]; } else { j = i = e - E; }
  float alpha = earr[e * Hh + h] / sbuf[i * Hh + h];
  int w = Hh * C;
  atomicAdd(&outacc[(size_t)i * w + t], Hf[(size_t)j * w + t] * alpha);
}

// Layers 1/2 epilogue: bias + ELU, emit f16 activations for the next WMMA GEMM.
__global__ void k_epilogue_elu_h16(const float* __restrict__ acc, const float* __restrict__ bias,
                                   _Float16* __restrict__ dst, int total) {
  int idx = blockIdx.x * blockDim.x + threadIdx.x;
  if (idx >= total) return;
  float v = acc[idx] + bias[idx & 127];
  v = (v > 0.f) ? v : (__expf(v) - 1.f);
  dst[idx] = (_Float16)v;
}

// Layer 3 epilogue (H=1 -> mean is identity): bias add, write h_node column block.
__global__ void k_epilogue_out(const float* __restrict__ acc, const float* __restrict__ bias,
                               float* __restrict__ out, int N, int coloff) {
  int idx = blockIdx.x * blockDim.x + threadIdx.x;
  if (idx >= N * 64) return;
  int n = idx >> 6, t = idx & 63;
  out[(size_t)n * 128 + coloff + t] = acc[idx] + bias[t];
}

__global__ void k_pool_acc(const float* __restrict__ hnode, const int* __restrict__ batch,
                           float* __restrict__ pool, float* __restrict__ cnt, int N) {
  int idx = blockIdx.x * blockDim.x + threadIdx.x;
  if (idx >= N * 128) return;
  int n = idx >> 7, t = idx & 127;
  int g = batch[n];
  atomicAdd(&pool[g * 128 + t], hnode[idx]);
  if (t == 0) atomicAdd(&cnt[g], 1.0f);
}

__global__ void k_pool_div(const float* __restrict__ pool, const float* __restrict__ cnt,
                           float* __restrict__ out) {
  int idx = blockIdx.x * blockDim.x + threadIdx.x;
  if (idx >= 64 * 128) return;
  out[idx] = pool[idx] / fmaxf(cnt[idx >> 7], 1.0f);
}

extern "C" void kernel_launch(void* const* d_in, const int* in_sizes, int n_in,
                              void* d_out, int out_size, void* d_ws, size_t ws_size,
                              hipStream_t stream) {
  (void)n_in; (void)out_size; (void)ws_size;
  const float* x     = (const float*)d_in[0];
  const int*   ei    = (const int*)d_in[1];
  const int*   batch = (const int*)d_in[2];
  const int N = in_sizes[0] / 5;
  const int E = in_sizes[1] / 2;
  const int* srcA = ei;       // edge_index[0]
  const int* dstA = ei + E;   // edge_index[1]

  // params flat order: f1,f2,f3,b1,b2,b3 x {W, att_src, att_dst, bias}
  const float* P[24];
  for (int k = 0; k < 24; ++k) P[k] = (const float*)d_in[3 + k];

  // workspace carve (256B aligned regions)
  char* wsb = (char*)d_ws;
  size_t off = 0;
  auto carve = [&](size_t bytes) -> void* {
    void* p = wsb + off; off += (bytes + 255) & ~(size_t)255; return p;
  };
  float*    bufH   = (float*)carve((size_t)N * 128 * 4);   // transformed features h
  float*    outacc = (float*)carve((size_t)N * 128 * 4);   // aggregation accumulator
  _Float16* h16    = (_Float16*)carve((size_t)N * 128 * 2);// f16 activations (GEMM A)
  _Float16* wpk    = (_Float16*)carve(64 * 1024);          // packed W fragments (GEMM B)
  float*    a_s    = (float*)carve((size_t)N * 4 * 4);
  float*    a_d    = (float*)carve((size_t)N * 4 * 4);
  float*    mbf    = (float*)carve((size_t)N * 4 * 4);
  float*    sbf    = (float*)carve((size_t)N * 4 * 4);
  float*    earr   = (float*)carve((size_t)(E + N) * 4 * 4);
  float*    pool   = (float*)carve(64 * 128 * 4);
  float*    cnt    = (float*)carve(64 * 4);

  float* hnode = (float*)d_out;              // [N,128]
  float* gpool = hnode + (size_t)N * 128;    // [64,128]

  const int TB = 256;
  auto g1 = [&](int total) { return dim3((unsigned)((total + TB - 1) / TB)); };

  auto run_gat = [&](const float* Wm, const float* atts, const float* attd, const float* bias,
                     int Hh, int C, int layer, const int* jidx, const int* iidx,
                     int c0, int c1, int c2, int coloff) {
    int Mw = Hh * C;
    if (layer == 1) {
      k_dense3<<<g1(N * 128), TB, 0, stream>>>(x, Wm, bufH, N, c0, c1, c2);
    } else {
      int kblks = 128 >> 5, mtiles = Mw >> 4;
      k_pack_w<<<g1(mtiles * kblks * 32), TB, 0, stream>>>(Wm, wpk, 128, Mw);
      int rowTiles = N / 16;                  // N=50000 -> 3125 exact tiles
      int waves = rowTiles * mtiles;
      k_gemm_wmma<<<g1(waves * 32), TB, 0, stream>>>(h16, wpk, bufH, rowTiles, 128, Mw);
    }
    k_node_prep<<<g1(N * Hh), TB, 0, stream>>>(bufH, atts, attd, a_s, a_d, mbf, sbf, N, Hh, C);
    k_zero_f32<<<g1(N * Mw), TB, 0, stream>>>(outacc, N * Mw);
    int tot = (E + N) * Hh;
    k_edge_logits<<<g1(tot), TB, 0, stream>>>(jidx, iidx, E, N, Hh, a_s, a_d, earr, mbf);
    k_edge_expsum<<<g1(tot), TB, 0, stream>>>(iidx, E, N, Hh, mbf, earr, sbf);
    dim3 blk((unsigned)Mw, (unsigned)(TB / Mw));
    dim3 grd((unsigned)((E + N + blk.y - 1) / blk.y));
    k_edge_aggregate<<<grd, blk, 0, stream>>>(jidx, iidx, E, N, Hh, C, bufH, earr, sbf, outacc);
    if (layer < 3) {
      k_epilogue_elu_h16<<<g1(N * 128), TB, 0, stream>>>(outacc, bias, h16, N * 128);
    } else {
      k_epilogue_out<<<g1(N * 64), TB, 0, stream>>>(outacc, bias, hnode, N, coloff);
    }
  };

  // forward branch: message j=src -> aggregate i=dst, input cols {0,1,3}
  run_gat(P[0],  P[1],  P[2],  P[3],  4, 32, 1, srcA, dstA, 0, 1, 3, 0);
  run_gat(P[4],  P[5],  P[6],  P[7],  4, 32, 2, srcA, dstA, 0, 0, 0, 0);
  run_gat(P[8],  P[9],  P[10], P[11], 1, 64, 3, srcA, dstA, 0, 0, 0, 0);
  // backward branch: message j=dst -> aggregate i=src, input cols {0,2,4}
  run_gat(P[12], P[13], P[14], P[15], 4, 32, 1, dstA, srcA, 0, 2, 4, 0);
  run_gat(P[16], P[17], P[18], P[19], 4, 32, 2, dstA, srcA, 0, 0, 0, 0);
  run_gat(P[20], P[21], P[22], P[23], 1, 64, 3, dstA, srcA, 0, 0, 0, 64);

  // graph mean-pool over h_node
  k_zero_f32<<<g1(64 * 128), TB, 0, stream>>>(pool, 64 * 128);
  k_zero_f32<<<1, 64, 0, stream>>>(cnt, 64);
  k_pool_acc<<<g1(N * 128), TB, 0, stream>>>(hnode, batch, pool, cnt, N);
  k_pool_div<<<g1(64 * 128), TB, 0, stream>>>(pool, cnt, gpool);
}
